// GNN_15831249453485
// MI455X (gfx1250) — matrix operand order repaired and checked by reading, not compiled, not run on hardware
//
#include <hip/hip_runtime.h>

// ---------------- static problem dims (match reference) ----------------
#define G_    64
#define NPG_  200
#define N_    12800
#define E_    64000
#define D_    1024
#define H_    3
#define C_    1024
#define HC_   3072
#define K1_   160
#define K2_   80
#define K3_   16

// ---------------- vector types ----------------
typedef __attribute__((ext_vector_type(16))) __bf16          v16bf;
typedef __attribute__((ext_vector_type(8)))  float           v8f;
typedef __attribute__((ext_vector_type(8)))  unsigned short  v8us;
typedef __attribute__((ext_vector_type(16))) unsigned short  v16us;
typedef __attribute__((ext_vector_type(4)))  unsigned int    v4u;
typedef __attribute__((ext_vector_type(8)))  int             v8i;
typedef __attribute__((ext_vector_type(4)))  int             v4i;

// float -> bf16 bits, round-to-nearest-even
__device__ __forceinline__ unsigned short f2bfu(float f) {
  unsigned u = __float_as_uint(f);
  unsigned r = u + 0x7FFFu + ((u >> 16) & 1u);
  return (unsigned short)(r >> 16);
}

// order-preserving float <-> uint encoding for atomicMax on floats
__device__ __forceinline__ unsigned fenc(float f) {
  unsigned u = __float_as_uint(f);
  return (u & 0x80000000u) ? ~u : (u | 0x80000000u);
}
__device__ __forceinline__ float fdec(unsigned u) {
  u = (u & 0x80000000u) ? (u & 0x7FFFFFFFu) : ~u;
  return __uint_as_float(u);
}

// ---------------- TDM availability ----------------
#if defined(__has_builtin)
#if __has_builtin(__builtin_amdgcn_tensor_load_to_lds) && \
    __has_builtin(__builtin_amdgcn_s_wait_tensorcnt)
#define USE_TDM 1
#endif
#endif
#ifndef USE_TDM
#define USE_TDM 0
#endif

// Compiler barrier: the TDM builtin's LDS write is invisible to LLVM alias
// analysis (the LDS address is an integer in the descriptor), so without
// this the ds_load fragment reads get folded to undef.
#define LDS_CLOBBER() asm volatile("" ::: "memory")

#define BM 128
#define BN 128
#define BK 32

#if USE_TDM
// Issue one TDM descriptor moving a 2D tile (tile_dim0=BK elems of 2B,
// tile_dim1=128 rows, row stride K elems) from global bf16 to LDS byte
// offset ldsOff.  D# bit layout per CDNA5 ISA §8.3/§8.4.
__device__ __forceinline__ void tdm_load_2d(unsigned ldsOff,
                                            const unsigned short* g,
                                            unsigned K) {
  unsigned long long ga = (unsigned long long)g;
  v4u g0;
  g0[0] = 1u;                                              // count=1, user D#
  g0[1] = ldsOff;                                          // lds_addr (bytes)
  g0[2] = (unsigned)ga;                                    // global_addr[31:0]
  g0[3] = (unsigned)((ga >> 32) & 0x01FFFFFFull) | (2u << 30); // [56:32]|type=2
  v8i g1;
  g1[0] = (int)(1u << 16);                                 // data_size=2B
  g1[1] = (int)((K & 0xFFFFu) << 16);                      // tensor_dim0 lo16
  g1[2] = (int)((K >> 16) | (128u << 16));                 // dim0 hi | dim1=128
  g1[3] = (int)((unsigned)BK << 16);                       // dim1 hi=0 | tile_dim0
  g1[4] = 128;                                             // tile_dim1=128
  g1[5] = (int)K;                                          // dim0_stride lo32
  g1[6] = 0;
  g1[7] = 0;
  v4i z4 = {0, 0, 0, 0};
#if __clang_major__ >= 23
  v8i z8 = {0, 0, 0, 0, 0, 0, 0, 0};
  __builtin_amdgcn_tensor_load_to_lds(g0, g1, z4, z4, z8, 0);
#else
  __builtin_amdgcn_tensor_load_to_lds(g0, g1, z4, z4, 0);
#endif
}
#endif

// =====================================================================
// bf16 WMMA GEMM:  C[M,N] = act(A @ B^T + bias)
//   A: bf16 [M,K] row-major (M multiple of 128)
//   B: bf16 [N,K] row-major, i.e. pre-transposed weights (N mult of 128)
//   K multiple of 32.  Output f32.
// Block: 256 threads (8 waves), tile 128x128, BK=32.
// TDM double-buffered LDS pipeline when available; sync copy fallback.
// =====================================================================
__global__ __launch_bounds__(256)
void k_gemm_bf16(const unsigned short* __restrict__ A,
                 const unsigned short* __restrict__ B,
                 const float* __restrict__ bias, float* __restrict__ Cmat,
                 int M, int Nn, int K, int act)
{
  // per buffer: A tile 128x32 (4096 us) + B tile 128x32 (4096 us) = 16 KB
  __shared__ __attribute__((aligned(128))) unsigned short smem[2 * 8192];

  const int tid  = threadIdx.x;
  const int bm   = blockIdx.y * BM;
  const int bn   = blockIdx.x * BN;
  const int wid  = tid >> 5;
  const int lane = tid & 31;
  const int wm   = (wid & 3) * 32;   // wave m offset
  const int wn   = (wid >> 2) * 64;  // wave n offset
  const int lr   = lane & 15;
  const int lh   = lane >> 4;

  v8f acc[2][4] = {};
  (void)M;

#if USE_TDM
  int cur = 0;
  if (wid == 0) {
    tdm_load_2d(0u,    A + (size_t)bm * K, (unsigned)K);
    tdm_load_2d(8192u, B + (size_t)bn * K, (unsigned)K);
    __builtin_amdgcn_s_wait_tensorcnt(0);
  }
  __syncthreads();
  asm volatile("" : : "r"(smem) : "memory");   // TDM wrote smem: force reload
  for (int k0 = 0; k0 < K; k0 += BK) {
    const int  nxt     = cur ^ 1;
    const bool hasNext = (k0 + BK) < K;
    if (wid == 0 && hasNext) {
      tdm_load_2d((unsigned)(nxt * 16384),        A + (size_t)bm * K + (k0 + BK), (unsigned)K);
      tdm_load_2d((unsigned)(nxt * 16384 + 8192), B + (size_t)bn * K + (k0 + BK), (unsigned)K);
    }
    const unsigned short* As = smem + cur * 8192;
    const unsigned short* Bs = As + 4096;

    v16bf af[2], bfr[4];
    for (int i = 0; i < 2; ++i) {
      int row = wm + i * 16 + lr;
      v8us lo = *(const v8us*)(As + row * BK + lh * 16);
      v8us hi = *(const v8us*)(As + row * BK + lh * 16 + 8);
      af[i] = __builtin_bit_cast(v16bf,
          __builtin_shufflevector(lo, hi, 0,1,2,3,4,5,6,7,8,9,10,11,12,13,14,15));
    }
    for (int j = 0; j < 4; ++j) {
      int col = wn + j * 16 + lr;
      bfr[j] = __builtin_bit_cast(v16bf, *(const v16us*)(Bs + col * BK + lh * 16));
    }
    for (int i = 0; i < 2; ++i)
      for (int j = 0; j < 4; ++j)
        acc[i][j] = __builtin_amdgcn_wmma_f32_16x16x32_bf16(
            false, af[i], false, bfr[j], (short)0, acc[i][j], false, false);

    if (wid == 0 && hasNext) __builtin_amdgcn_s_wait_tensorcnt(0);
    __syncthreads();
    LDS_CLOBBER();                              // next buffer now valid
    cur = nxt;
  }
#else
  for (int k0 = 0; k0 < K; k0 += BK) {
    unsigned short* As = smem;
    unsigned short* Bs = smem + 4096;
    // cooperative staging: 512 16-byte chunks per tile, 2 per thread
    for (int i = 0; i < 2; ++i) {
      int c   = tid + i * 256;
      int row = c >> 2;
      int co  = (c & 3) * 8;
      *(v8us*)(As + row * BK + co) = *(const v8us*)(A + (size_t)(bm + row) * K + k0 + co);
      *(v8us*)(Bs + row * BK + co) = *(const v8us*)(B + (size_t)(bn + row) * K + k0 + co);
    }
    __syncthreads();
    v16bf af[2], bfr[4];
    for (int i = 0; i < 2; ++i) {
      int row = wm + i * 16 + lr;
      v8us lo = *(const v8us*)(As + row * BK + lh * 16);
      v8us hi = *(const v8us*)(As + row * BK + lh * 16 + 8);
      af[i] = __builtin_bit_cast(v16bf,
          __builtin_shufflevector(lo, hi, 0,1,2,3,4,5,6,7,8,9,10,11,12,13,14,15));
    }
    for (int j = 0; j < 4; ++j) {
      int col = wn + j * 16 + lr;
      bfr[j] = __builtin_bit_cast(v16bf, *(const v16us*)(Bs + col * BK + lh * 16));
    }
    for (int i = 0; i < 2; ++i)
      for (int j = 0; j < 4; ++j)
        acc[i][j] = __builtin_amdgcn_wmma_f32_16x16x32_bf16(
            false, af[i], false, bfr[j], (short)0, acc[i][j], false, false);
    __syncthreads();
  }
#endif

  // store: M,N multiples of 128 -> no guards
  for (int i = 0; i < 2; ++i)
    for (int j = 0; j < 4; ++j) {
      int nC = bn + wn + j * 16 + lr;
      float bv = bias ? bias[nC] : 0.f;
      for (int r = 0; r < 8; ++r) {
        int mC = bm + wm + i * 16 + lh * 8 + r;
        float v = acc[i][j][r] + bv;
        if (act == 1) v = v > 0.f ? v : 0.f;
        Cmat[(size_t)mC * Nn + nC] = v;
      }
    }
}

// =====================================================================
// conversion / transpose prep kernels
// =====================================================================
// f32 [rows,K] -> bf16 [rowsPad,K]; rows beyond rowsValid zero-filled
__global__ void k_cvt(const float* __restrict__ in, unsigned short* out,
                      int rowsValid, int K, long long total) {
  long long idx = (long long)blockIdx.x * 256 + threadIdx.x;
  if (idx >= total) return;
  long long row = idx / K;
  out[idx] = (row < rowsValid) ? f2bfu(in[idx]) : (unsigned short)0;
}
// f32 [K,Nmat] -> bf16 [Nmat,K] (transpose + convert; for weights)
__global__ void k_tcvt(const float* __restrict__ in, unsigned short* out,
                       int K, int Nmat, long long total) {
  long long idx = (long long)blockIdx.x * 256 + threadIdx.x;
  if (idx >= total) return;
  int n = (int)(idx / K), k = (int)(idx % K);
  out[idx] = f2bfu(in[(size_t)k * Nmat + n]);
}

// =====================================================================
// small utility kernels
// =====================================================================
__global__ void k_fill_f32(float* p, float v, long long n) {
  long long i = (long long)blockIdx.x * 256 + threadIdx.x;
  if (i < n) p[i] = v;
}
__global__ void k_fill_u32(unsigned* p, unsigned v, long long n) {
  long long i = (long long)blockIdx.x * 256 + threadIdx.x;
  if (i < n) p[i] = v;
}
__global__ void k_fill_i32(int* p, int v, long long n) {
  long long i = (long long)blockIdx.x * 256 + threadIdx.x;
  if (i < n) p[i] = v;
}
__global__ void k_init_rowbias(float* out, const float* __restrict__ b, long long total) {
  long long i = (long long)blockIdx.x * 256 + threadIdx.x;
  if (i < total) out[i] = b[i % HC_];
}
__global__ void k_init_edges(const int* __restrict__ eidx, int* s, int* d, int* m) {
  int e = blockIdx.x * 256 + threadIdx.x;
  if (e >= E_) return;
  s[e] = eidx[e]; d[e] = eidx[E_ + e]; m[e] = 1;
}

// =====================================================================
// GAT attention
// =====================================================================
__global__ __launch_bounds__(256)
void k_attn_coef(const float* __restrict__ xw, const float* __restrict__ att_s,
                 const float* __restrict__ att_d, float* a_s, float* a_d, int n)
{
  int gw   = (blockIdx.x * 256 + threadIdx.x) >> 5;
  int lane = threadIdx.x & 31;
  if (gw >= n * H_) return;
  int node = gw / H_, h = gw % H_;
  const float* row = xw + (size_t)node * HC_ + h * C_;
  const float* ws_ = att_s + h * C_;
  const float* wd_ = att_d + h * C_;
  float ss = 0.f, sd = 0.f;
  for (int c = lane; c < C_; c += 32) {
    float v = row[c];
    ss += v * ws_[c];
    sd += v * wd_[c];
  }
  for (int off = 16; off; off >>= 1) {
    ss += __shfl_down(ss, off, 32);
    sd += __shfl_down(sd, off, 32);
  }
  if (lane == 0) { a_s[gw] = ss; a_d[gw] = sd; }
}

__device__ __forceinline__ float lrelu(float x) { return x > 0.f ? x : 0.2f * x; }

__global__ void k_edge_max(const int* __restrict__ src, const int* __restrict__ dst,
                           const int* __restrict__ msk, const float* __restrict__ a_s,
                           const float* __restrict__ a_d, unsigned* mx, int n)
{
  int e = blockIdx.x * 256 + threadIdx.x;
  if (e >= E_ + n) return;
  int s, d;
  if (e < E_) { if (!msk[e]) return; s = src[e]; d = dst[e]; }
  else { s = d = e - E_; }
  for (int h = 0; h < H_; ++h) {
    float l = lrelu(a_s[s * H_ + h] + a_d[d * H_ + h]);
    atomicMax(&mx[d * H_ + h], fenc(l));
  }
}

__global__ void k_edge_exp(const int* __restrict__ src, const int* __restrict__ dst,
                           const int* __restrict__ msk, const float* __restrict__ a_s,
                           const float* __restrict__ a_d, const unsigned* __restrict__ mx,
                           float* exb, float* den, int n)
{
  int e = blockIdx.x * 256 + threadIdx.x;
  if (e >= E_ + n) return;
  int s, d;
  if (e < E_) {
    if (!msk[e]) { for (int h = 0; h < H_; ++h) exb[e * H_ + h] = 0.f; return; }
    s = src[e]; d = dst[e];
  } else { s = d = e - E_; }
  for (int h = 0; h < H_; ++h) {
    float l = lrelu(a_s[s * H_ + h] + a_d[d * H_ + h]);
    float v = __expf(l - fdec(mx[d * H_ + h]));
    exb[e * H_ + h] = v;
    atomicAdd(&den[d * H_ + h], v);
  }
}

// one wave per (edge, head): scatter alpha * xw[src] into out[dst]
__global__ __launch_bounds__(256)
void k_aggregate(const int* __restrict__ src, const int* __restrict__ dst,
                 const int* __restrict__ msk, const float* __restrict__ exb,
                 const float* __restrict__ den, const float* __restrict__ xw,
                 float* out, int n)
{
  int gw   = (blockIdx.x * 256 + threadIdx.x) >> 5;
  int lane = threadIdx.x & 31;
  int tot  = (E_ + n) * H_;
  if (gw >= tot) return;
  int e = gw / H_, h = gw % H_;
  int s, d;
  if (e < E_) { if (!msk[e]) return; s = src[e]; d = dst[e]; }
  else { s = d = e - E_; }
  float alpha = exb[e * H_ + h] / (den[d * H_ + h] + 1e-16f);
  const float* xr = xw + (size_t)s * HC_ + h * C_;
  float*      orw = out + (size_t)d * HC_ + h * C_;
  for (int c = lane; c < C_; c += 32)
    atomicAdd(&orw[c], alpha * xr[c]);
}

// =====================================================================
// TopK pooling
// =====================================================================
__global__ void k_pnorm(const float* __restrict__ p, float* out) {
  __shared__ float red[256];
  float s = 0.f;
  for (int c = threadIdx.x; c < C_; c += 256) s += p[c] * p[c];
  red[threadIdx.x] = s; __syncthreads();
  for (int o = 128; o; o >>= 1) {
    if (threadIdx.x < o) red[threadIdx.x] += red[threadIdx.x + o];
    __syncthreads();
  }
  if (threadIdx.x == 0) out[0] = sqrtf(red[0]);
}

__global__ __launch_bounds__(256)
void k_score(const float* __restrict__ hm, const float* __restrict__ p,
             const float* __restrict__ pn, float* score, int n)
{
  int gw   = (blockIdx.x * 256 + threadIdx.x) >> 5;
  int lane = threadIdx.x & 31;
  if (gw >= n) return;
  const float* row = hm + (size_t)gw * C_;
  float s = 0.f;
  for (int c = lane; c < C_; c += 32) s += row[c] * p[c];
  for (int off = 16; off; off >>= 1) s += __shfl_down(s, off, 32);
  if (lane == 0) score[gw] = tanhf(s / pn[0]);
}

// one block per graph: bitonic sort 256 (score desc, idx asc) in LDS
__global__ __launch_bounds__(256)
void k_topk(const float* __restrict__ score, float* vals, int* perm, int* newid,
            int n_in, int k)
{
  __shared__ float ss[256];
  __shared__ int   si[256];
  int g = blockIdx.x, t = threadIdx.x;
  ss[t] = (t < n_in) ? score[g * n_in + t] : -3.0e38f;
  si[t] = t;
  __syncthreads();
  for (int ksz = 2; ksz <= 256; ksz <<= 1) {
    for (int j = ksz >> 1; j > 0; j >>= 1) {
      int ixj = t ^ j;
      if (ixj > t) {
        float s1 = ss[t], s2 = ss[ixj];
        int   i1 = si[t], i2 = si[ixj];
        bool aFirst = (s1 > s2) || (s1 == s2 && i1 < i2);
        bool up = ((t & ksz) == 0);
        if (up ? !aFirst : aFirst) {
          ss[t] = s2; ss[ixj] = s1; si[t] = i2; si[ixj] = i1;
        }
      }
      __syncthreads();
    }
  }
  if (t < k) {
    vals[g * k + t] = ss[t];
    perm[g * k + t] = g * n_in + si[t];
    newid[g * n_in + si[t]] = g * k + t;
  }
}

__global__ void k_gather(const float* __restrict__ hin, const int* __restrict__ perm,
                         const float* __restrict__ vals, float* hout, int nout)
{
  long long idx = (long long)blockIdx.x * 256 + threadIdx.x;
  long long tot = (long long)nout * C_;
  if (idx >= tot) return;
  int i = (int)(idx / C_), c = (int)(idx % C_);
  hout[idx] = hin[(size_t)perm[i] * C_ + c] * vals[i];
}

__global__ void k_remap(const int* __restrict__ newid, int* s, int* d, int* m) {
  int e = blockIdx.x * 256 + threadIdx.x;
  if (e >= E_) return;
  int mm = m[e];
  int ns = newid[s[e]], nd = newid[d[e]];
  int nm = (mm && ns >= 0 && nd >= 0) ? 1 : 0;
  s[e] = nm ? ns : 0;
  d[e] = nm ? nd : 0;
  m[e] = nm;
}

// =====================================================================
// readout (max & mean per graph, accumulated into z) and final head
// =====================================================================
__global__ void k_readout(const float* __restrict__ x, float* z, int k) {
  int idx = blockIdx.x * 256 + threadIdx.x;
  if (idx >= G_ * C_) return;
  int g = idx / C_, c = idx % C_;
  const float* base = x + (size_t)g * k * C_ + c;
  float mx = -3.0e38f, sm = 0.f;
  for (int j = 0; j < k; ++j) {
    float v = base[(size_t)j * C_];
    mx = v > mx ? v : mx;
    sm += v;
  }
  z[(size_t)g * 2 * C_ + c]      += mx;
  z[(size_t)g * 2 * C_ + C_ + c] += sm * (1.0f / (float)k);
}

__global__ void k_head(const float* __restrict__ y1, const float* __restrict__ Wl2,
                       const float* __restrict__ bl2, float* out) {
  int t = blockIdx.x * 128 + threadIdx.x;
  if (t >= G_ * 2) return;
  int g = t >> 1, o = t & 1;
  const float* row = y1 + (size_t)g * 1024;
  float s = bl2[o];
  for (int kk = 0; kk < 1024; ++kk) s += row[kk] * Wl2[kk * 2 + o];
  out[t] = s;
}

// =====================================================================
// host driver
// =====================================================================
static inline dim3 b1d(long long n, int t) { return dim3((unsigned)((n + t - 1) / t)); }

static void run_gemm(const unsigned short* A, const unsigned short* B,
                     const float* bias, float* Cp,
                     int M, int Nn, int K, int act, hipStream_t s) {
  dim3 grid((Nn + BN - 1) / BN, (M + BM - 1) / BM);
  k_gemm_bf16<<<grid, dim3(256), 0, s>>>(A, B, bias, Cp, M, Nn, K, act);
}

extern "C" void kernel_launch(void* const* d_in, const int* in_sizes, int n_in,
                              void* d_out, int out_size, void* d_ws, size_t ws_size,
                              hipStream_t stream)
{
  (void)in_sizes; (void)n_in; (void)out_size; (void)ws_size;

  const float* x    = (const float*)d_in[0];
  const int*   eidx = (const int*)  d_in[2];
  const float* W[3]  = {(const float*)d_in[4],  (const float*)d_in[11], (const float*)d_in[18]};
  const float* AS[3] = {(const float*)d_in[5],  (const float*)d_in[12], (const float*)d_in[19]};
  const float* AD[3] = {(const float*)d_in[6],  (const float*)d_in[13], (const float*)d_in[20]};
  const float* BB[3] = {(const float*)d_in[7],  (const float*)d_in[14], (const float*)d_in[21]};
  const float* WT[3] = {(const float*)d_in[8],  (const float*)d_in[15], (const float*)d_in[22]};
  const float* BT[3] = {(const float*)d_in[9],  (const float*)d_in[16], (const float*)d_in[23]};
  const float* P[3]  = {(const float*)d_in[10], (const float*)d_in[17], (const float*)d_in[24]};
  const float* Wl1 = (const float*)d_in[25];
  const float* bl1 = (const float*)d_in[26];
  const float* Wl2 = (const float*)d_in[27];
  const float* bl2 = (const float*)d_in[28];

  // ---- workspace carve-up ----
  char* ws = (char*)d_ws;
  size_t off = 0;
  auto carve = [&](size_t bytes) -> char* {
    char* p = ws + off;
    off += (bytes + 255) & ~(size_t)255;
    return p;
  };
  float*          Abig  = (float*)carve((size_t)N_ * HC_ * 4);     // xw / h
  float*          Bbig  = (float*)carve((size_t)N_ * HC_ * 4);     // aggregation
  float*          Hs    = (float*)carve((size_t)G_ * K1_ * C_ * 4);
  unsigned short* Abf   = (unsigned short*)carve((size_t)N_ * HC_ * 2); // bf16 A scratch
  unsigned short* Wbf[3], *WTbf[3];
  for (int L = 0; L < 3; ++L) {
    Wbf[L]  = (unsigned short*)carve((size_t)HC_ * 1024 * 2);      // [3072,1024]
    WTbf[L] = (unsigned short*)carve((size_t)1024 * HC_ * 2);      // [1024,3072]
  }
  unsigned short* Wl1bf = (unsigned short*)carve((size_t)1024 * 2048 * 2); // [1024,2048]
  float*    a_s   = (float*)   carve((size_t)N_ * H_ * 4);
  float*    a_d   = (float*)   carve((size_t)N_ * H_ * 4);
  unsigned* mx    = (unsigned*)carve((size_t)N_ * H_ * 4);
  float*    den   = (float*)   carve((size_t)N_ * H_ * 4);
  float*    exb   = (float*)   carve((size_t)(E_ + N_) * H_ * 4);
  float*    score = (float*)   carve((size_t)N_ * 4);
  float*    vals  = (float*)   carve((size_t)G_ * K1_ * 4);
  int*      perm  = (int*)     carve((size_t)G_ * K1_ * 4);
  int*      newid = (int*)     carve((size_t)N_ * 4);
  int*      srcb  = (int*)     carve((size_t)E_ * 4);
  int*      dstb  = (int*)     carve((size_t)E_ * 4);
  int*      mskb  = (int*)     carve((size_t)E_ * 4);
  float*    pnrm  = (float*)   carve(256);
  float*    z     = (float*)   carve((size_t)G_ * 2 * C_ * 4);
  float*    y1    = (float*)   carve((size_t)128 * 1024 * 4);

  const int nNodes[3] = {N_, G_ * K1_, G_ * K2_};
  const int kpg[3]    = {K1_, K2_, K3_};
  const int npgIn[3]  = {NPG_, K1_, K2_};

  // ---- one-time per call: convert+transpose all weight matrices to bf16 ----
  for (int L = 0; L < 3; ++L) {
    k_tcvt<<<b1d((long long)HC_ * 1024, 256), 256, 0, stream>>>(W[L],  Wbf[L],  1024, HC_,  (long long)HC_ * 1024);
    k_tcvt<<<b1d((long long)HC_ * 1024, 256), 256, 0, stream>>>(WT[L], WTbf[L], HC_,  1024, (long long)1024 * HC_);
  }
  k_tcvt<<<b1d((long long)1024 * 2048, 256), 256, 0, stream>>>(Wl1, Wl1bf, 2048, 1024, (long long)1024 * 2048);

  k_fill_f32 <<<b1d((long long)G_ * 2 * C_, 256), 256, 0, stream>>>(z, 0.f, (long long)G_ * 2 * C_);
  k_init_edges<<<b1d(E_, 256), 256, 0, stream>>>(eidx, srcb, dstb, mskb);

  for (int L = 0; L < 3; ++L) {
    const int n = nNodes[L];
    const float* xin = (L == 0) ? x : Hs;

    // xw = xin @ W  -> Abig [n, 3072]
    k_cvt<<<b1d((long long)n * 1024, 256), 256, 0, stream>>>(xin, Abf, n, 1024, (long long)n * 1024);
    run_gemm(Abf, Wbf[L], nullptr, Abig, n, HC_, 1024, 0, stream);

    // attention coefficients
    k_attn_coef<<<b1d((long long)n * H_ * 32, 256), 256, 0, stream>>>(Abig, AS[L], AD[L], a_s, a_d, n);

    // segment softmax over destinations (self-loops appended)
    k_fill_u32<<<b1d((long long)n * H_, 256), 256, 0, stream>>>(mx, 0u, (long long)n * H_);
    k_fill_f32<<<b1d((long long)n * H_, 256), 256, 0, stream>>>(den, 0.f, (long long)n * H_);
    k_edge_max<<<b1d(E_ + n, 256), 256, 0, stream>>>(srcb, dstb, mskb, a_s, a_d, mx, n);
    k_edge_exp<<<b1d(E_ + n, 256), 256, 0, stream>>>(srcb, dstb, mskb, a_s, a_d, mx, exb, den, n);

    // weighted aggregation into Bbig (init with bias b)
    k_init_rowbias<<<b1d((long long)n * HC_, 256), 256, 0, stream>>>(Bbig, BB[L], (long long)n * HC_);
    k_aggregate<<<b1d((long long)(E_ + n) * H_ * 32, 256), 256, 0, stream>>>(
        srcb, dstb, mskb, exb, den, Abig, Bbig, n);

    // h = agg @ Wt + bt  -> Abig (front [n,1024])
    k_cvt<<<b1d((long long)n * HC_, 256), 256, 0, stream>>>(Bbig, Abf, n, HC_, (long long)n * HC_);
    run_gemm(Abf, WTbf[L], BT[L], Abig, n, C_, HC_, 0, stream);

    // TopK pooling
    k_pnorm<<<1, 256, 0, stream>>>(P[L], pnrm);
    k_score<<<b1d((long long)n * 32, 256), 256, 0, stream>>>(Abig, P[L], pnrm, score, n);
    k_fill_i32<<<b1d(n, 256), 256, 0, stream>>>(newid, -1, n);
    k_topk<<<G_, 256, 0, stream>>>(score, vals, perm, newid, npgIn[L], kpg[L]);
    k_gather<<<b1d((long long)G_ * kpg[L] * C_, 256), 256, 0, stream>>>(
        Abig, perm, vals, Hs, G_ * kpg[L]);
    k_remap<<<b1d(E_, 256), 256, 0, stream>>>(newid, srcb, dstb, mskb);

    // readout accumulates into z
    k_readout<<<b1d((long long)G_ * C_, 256), 256, 0, stream>>>(Hs, z, kpg[L]);
  }

  // MLP head: y1 = relu(z @ Wl1 + bl1)  (M padded 64->128 with zero rows)
  k_cvt<<<b1d(128LL * 2048, 256), 256, 0, stream>>>(z, Abf, G_, 2048, 128LL * 2048);
  run_gemm(Abf, Wl1bf, bl1, y1, 128, 1024, 2048, 1, stream);
  k_head<<<b1d(G_ * 2, 128), 128, 0, stream>>>(y1, Wl2, bl2, (float*)d_out);
}
